// ModulatedConv2d_7782480740973
// MI455X (gfx1250) — matrix operand order repaired and checked by reading, not compiled, not run on hardware
//
#include <hip/hip_runtime.h>

// ---------------------------------------------------------------------------
// ModulatedConv2d (StyleGAN2) for gfx1250 via bf16 WMMA implicit GEMM.
// B=16, Cin=Cout=256, H=W=64, K=3, S=512.
//   Stage A : s[b,ci] = style . mod_weight^T + mod_bias
//   Stage B : demod + modulated weights -> bf16 ws  [b][co][tap][ci]
//   Stage B2: x f32 -> bf16 ws (one-time conversion)
//   Stage C : implicit GEMM; B staged as 256-pixel halo in LDS (reused by all
//             9 taps); A tiles moved by the Tensor Data Mover (TDM) with LDS
//             pad, double-buffered against v_wmma_f32_16x16x32_bf16 compute.
// ---------------------------------------------------------------------------

#define BATCH 16
#define CIN   256
#define COUT  256
#define IMH   64
#define IMW   64
#define NPIX  (IMH * IMW)      // 4096
#define STY   512
#define NTAP  9

#define BT_M  128              // block tile rows (cout)
#define BT_N  128              // block tile cols (pixels)
#define KSTEP 32               // WMMA K per step (bf16)
#define LDSA_STRIDE 40         // shorts per A row (80B: conflict-free frags)
#define LDSB_STRIDE 40         // shorts per B halo pixel
#define HALO  256              // halo pixels per ci-chunk (n0-64 .. n0+191)
#define KROW  (NTAP * CIN)     // 2304: wmod elements per (b,co) row

typedef __bf16 bf16_t;
typedef __attribute__((ext_vector_type(16))) bf16_t bf16x16;
typedef __attribute__((ext_vector_type(8)))  float  f32x8;
typedef __attribute__((ext_vector_type(4)))  unsigned int u32x4;
typedef __attribute__((ext_vector_type(8)))  int i32x8;
typedef __attribute__((ext_vector_type(4)))  int i32x4;

struct Bits256 { u32x4 lo, hi; };   // 32 bytes == bf16x16

// float -> bf16 RNE (raw 16-bit pattern)
__device__ __forceinline__ unsigned short f2bf(float f) {
    union { float f; unsigned int u; } c;
    c.f = f;
    unsigned int u = c.u;
    u += 0x7FFFu + ((u >> 16) & 1u);
    return (unsigned short)(u >> 16);
}
// two f32 bit patterns -> packed bf16 pair
__device__ __forceinline__ unsigned int pack2bf(unsigned int lo, unsigned int hi) {
    unsigned int l = (lo + 0x7FFFu + ((lo >> 16) & 1u)) >> 16;
    unsigned int h = (hi + 0x7FFFu + ((hi >> 16) & 1u)) & 0xFFFF0000u;
    return l | h;
}

// ---------------------------------------------------------------------------
// TDM: DMA a 128x32 bf16 tile (row stride 2304 elems) global -> LDS, padding
// 16B after each 64B row so the LDS row stride is 80B (= LDSA_STRIDE shorts).
// Descriptor per CDNA5 ISA section 8 (groups 0/1; 2D tile so groups 2/3 = 0).
// This toolchain's builtin takes 6 args (extra i32x8 group + cpol).
// ---------------------------------------------------------------------------
__device__ __forceinline__ void tdm_load_A(unsigned lds_off,
                                           const unsigned short* gsrc) {
    const unsigned long long ga = (unsigned long long)gsrc;
    u32x4 g0;
    g0[0] = 1u;                                    // count=1, user mode
    g0[1] = lds_off;                               // lds_addr (bytes)
    g0[2] = (unsigned)(ga & 0xFFFFFFFFu);          // global_addr[31:0]
    g0[3] = (unsigned)((ga >> 32) & 0x1FFFFFFu)    // global_addr[56:32]
          | (2u << 30);                            // type = 2 (image)
    i32x8 g1;
    g1[0] = (int)((1u << 16)      // data_size = 1 -> 2 bytes
                | (1u << 20)      // pad_enable
                | (3u << 22)      // pad_interval = 3 -> every 16 DWORDs (64B)
                | (3u << 25));    // pad_amount   = 3 -> 4 DWORDs (16B)
    g1[1] = (int)((unsigned)KROW << 16);   // tensor_dim0[15:0] in bits 63:48
    g1[2] = (int)(128u << 16);             // tensor_dim1[15:0] in bits 95:80
    g1[3] = (int)(32u << 16);              // tile_dim0 = 32    in bits 127:112
    g1[4] = 128;                           // tile_dim1 = 128   in bits 143:128
    g1[5] = KROW;                          // tensor_dim0_stride[31:0]
    g1[6] = 0;
    g1[7] = 0;
    const i32x4 z4 = {0, 0, 0, 0};
    const i32x8 z8 = {0, 0, 0, 0, 0, 0, 0, 0};
    __builtin_amdgcn_tensor_load_to_lds(g0, g1, z4, z4, z8, 0);
}

// ---------------------------------------------------------------------------
// Stage A: s[b, ci]
// ---------------------------------------------------------------------------
__global__ __launch_bounds__(256)
void style_mod_kernel(const float* __restrict__ style,
                      const float* __restrict__ mod_weight,
                      const float* __restrict__ mod_bias,
                      float* __restrict__ s_out) {
    __shared__ float st[STY];
    const int b = blockIdx.x;
    const int t = threadIdx.x;
    st[t]       = style[b * STY + t];
    st[t + 256] = style[b * STY + t + 256];
    __syncthreads();

    const float* wr = mod_weight + t * STY;
    float acc = mod_bias[t];
#pragma unroll 4
    for (int k = 0; k < STY; ++k)
        acc += st[k] * wr[k];
    s_out[b * CIN + t] = acc;
}

// ---------------------------------------------------------------------------
// Stage B: demod + modulated bf16 weights, layout [b][co][tap][ci]
// ---------------------------------------------------------------------------
__global__ __launch_bounds__(256)
void demod_wmod_kernel(const float* __restrict__ weight,   // [COUT][CIN][9]
                       const float* __restrict__ s_in,     // [B][CIN]
                       unsigned short* __restrict__ wmod) {
    __shared__ float red[256];
    const int co = blockIdx.x;
    const int b  = blockIdx.y;
    const int ci = threadIdx.x;

    const float sv = s_in[b * CIN + ci];
    const int wbase = (co * CIN + ci) * NTAP;

    float wm[NTAP];
    float part = 0.f;
#pragma unroll
    for (int tp = 0; tp < NTAP; ++tp) {
        float m = weight[wbase + tp] * sv;
        wm[tp] = m;
        part += m * m;
    }
    red[ci] = part;
    __syncthreads();
#pragma unroll
    for (int s = 128; s > 0; s >>= 1) {
        if (ci < s) red[ci] += red[ci + s];
        __syncthreads();
    }
    const float demod = rsqrtf(red[0] + 1e-8f);

    const int obase = ((b * COUT + co) * NTAP) * CIN + ci;
#pragma unroll
    for (int tp = 0; tp < NTAP; ++tp)
        wmod[obase + tp * CIN] = f2bf(wm[tp] * demod);
}

// ---------------------------------------------------------------------------
// Stage B2: x f32 -> bf16 (one time). 8 elements / thread.
// ---------------------------------------------------------------------------
__global__ __launch_bounds__(256)
void x_to_bf16_kernel(const float* __restrict__ xg,
                      unsigned short* __restrict__ xbf) {
    const int gid = blockIdx.x * 256 + threadIdx.x;
    const u32x4* src = (const u32x4*)xg + gid * 2;
    u32x4 a = src[0];
    u32x4 b = src[1];
    u32x4 o;
    o[0] = pack2bf(a[0], a[1]);
    o[1] = pack2bf(a[2], a[3]);
    o[2] = pack2bf(b[0], b[1]);
    o[3] = pack2bf(b[2], b[3]);
    ((u32x4*)xbf)[gid] = o;
}

// ---------------------------------------------------------------------------
// Stage C: implicit GEMM conv.
// grid = (32, 2, 16), block = 256 (8 waves).
// wave (wm 0..3, wn 0..1) owns 32(M) x 64(N): 2x4 wmma accumulators.
// A tiles: TDM double-buffered.  B: 256-pixel halo reused by 9 taps.
// ---------------------------------------------------------------------------
__global__ __launch_bounds__(256)
void conv_wmma_kernel(const unsigned short* __restrict__ xbf,  // [B][CIN][4096] bf16
                      const unsigned short* __restrict__ wmod, // [B][COUT][9][CIN] bf16
                      const float* __restrict__ bias,          // [COUT]
                      float* __restrict__ out) {               // [B][COUT][4096]
    __shared__ unsigned short Ald[2][BT_M * LDSA_STRIDE];  // 2 x 10.0 KB
    __shared__ unsigned short Bld[HALO * LDSB_STRIDE];     // 20.0 KB
    __shared__ unsigned short Zrow[16];                    // zero row for pad

    const int t     = threadIdx.x;
    const int lane  = t & 31;
    const int wv    = t >> 5;
    const int wm    = wv & 3;
    const int wn    = wv >> 2;
    const int khalf = lane >> 4;
    const int l16   = lane & 15;

    const int b   = blockIdx.z;
    const int co0 = blockIdx.y * BT_M;
    const int n0  = blockIdx.x * BT_N;

    if (t < 16) Zrow[t] = 0;

    // per-lane B fragment geometry
    int pixc[4], xcoord[4];
#pragma unroll
    for (int c = 0; c < 4; ++c) {
        pixc[c]   = wn * 64 + c * 16 + l16;   // 0..127
        xcoord[c] = pixc[c] & 63;             // image x (n0 multiple of 64)
    }

    f32x8 acc[2][4];
#pragma unroll
    for (int r = 0; r < 2; ++r)
#pragma unroll
        for (int c = 0; c < 4; ++c)
#pragma unroll
            for (int v = 0; v < 8; ++v)
                acc[r][c][v] = 0.f;

    const unsigned short* xb_base = xbf + ((b * CIN) << 12);
    const unsigned short* wtile0  = wmod + (b * COUT + co0) * KROW;
    const unsigned ldsA[2] = {
        (unsigned)(unsigned long long)&Ald[0][0],
        (unsigned)(unsigned long long)&Ald[1][0]
    };

    // prologue: DMA the first A tile (ci0=0, tap=0) into buffer 0
    if (wv == 0)
        tdm_load_A(ldsA[0], wtile0);

#pragma unroll 1
    for (int ci0 = 0; ci0 < CIN; ci0 += KSTEP) {
        const int cidx = ci0 >> 5;         // 0..7

        __syncthreads();   // all waves done reading Bld (previous chunk)

        // ---- stage B halo: 256 pixels x 32 ci bf16 (u32 = 2 pixels of a ci)
#pragma unroll
        for (int k = 0; k < 16; ++k) {
            const int j   = k * 256 + t;   // 0..4095
            const int pr  = j & 127;       // pixel-pair index
            const int ciL = j >> 7;        // 0..31
            const int hp  = pr * 2;
            const int Pg  = n0 - 64 + hp;  // flattened global pixel
            unsigned int v = 0u;
            if ((unsigned)Pg < (unsigned)NPIX)
                v = *(const unsigned int*)(xb_base + ((ci0 + ciL) << 12) + Pg);
            Bld[hp * LDSB_STRIDE + ciL]       = (unsigned short)v;
            Bld[(hp + 1) * LDSB_STRIDE + ciL] = (unsigned short)(v >> 16);
        }

        // ---- 9 taps reuse the halo; A tiles arrive via TDM double-buffer
#pragma unroll
        for (int tap = 0; tap < NTAP; ++tap) {
            const int dy = tap / 3 - 1;
            const int dx = tap % 3 - 1;
            const int pb = (cidx * NTAP + tap) & 1;   // current A buffer

            if (wv == 0)
                __builtin_amdgcn_s_wait_tensorcnt(0);  // A tile landed
            __syncthreads();  // A+B visible; prev-prev A buffer free

            // issue next A tile into the alternate buffer (overlaps compute)
            if (wv == 0) {
                int ntap = tap + 1, nci0 = ci0;
                if (ntap == NTAP) { ntap = 0; nci0 += KSTEP; }
                if (nci0 < CIN)
                    tdm_load_A(ldsA[pb ^ 1],
                               wtile0 + (ntap << 8) + nci0);
            }

            // ---- A fragments (contiguous k = 16*khalf + s)
            bf16x16 afr[2], bfr[4];
#pragma unroll
            for (int r = 0; r < 2; ++r) {
                const unsigned short* ap =
                    &Ald[pb][(wm * 32 + r * 16 + l16) * LDSA_STRIDE];
                Bits256 bits;
                bits.lo = *(const u32x4*)(ap + khalf * 16);      // K 0-7 | 16-23
                bits.hi = *(const u32x4*)(ap + khalf * 16 + 8);  // K 8-15|24-31
                afr[r] = __builtin_bit_cast(bf16x16, bits);
            }
            // ---- B fragments: shifted pixel from halo; edge lanes -> Zrow
#pragma unroll
            for (int c = 0; c < 4; ++c) {
                const int hp = pixc[c] + 64 + dy * 64 + dx;
                const bool valid =
                    (dx == 0) | ((unsigned)(xcoord[c] + dx) < (unsigned)IMW);
                const unsigned short* bp =
                    valid ? &Bld[hp * LDSB_STRIDE + khalf * 16] : &Zrow[0];
                Bits256 bits;
                bits.lo = ((const u32x4*)bp)[0];
                bits.hi = ((const u32x4*)bp)[1];
                bfr[c] = __builtin_bit_cast(bf16x16, bits);
            }
#pragma unroll
            for (int r = 0; r < 2; ++r)
#pragma unroll
                for (int c = 0; c < 4; ++c)
                    acc[r][c] = __builtin_amdgcn_wmma_f32_16x16x32_bf16(
                        false, afr[r], false, bfr[c],
                        (short)0, acc[r][c], false, false);
        }
    }

    // ---- epilogue: C layout VGPR v -> M = v + 8*khalf, N = l16
#pragma unroll
    for (int r = 0; r < 2; ++r)
#pragma unroll
        for (int c = 0; c < 4; ++c) {
            const int coBase = co0 + wm * 32 + r * 16 + khalf * 8;
            const int P      = n0 + wn * 64 + c * 16 + l16;
#pragma unroll
            for (int v = 0; v < 8; ++v) {
                const int co = coBase + v;
                out[((b * COUT + co) << 12) + P] = acc[r][c][v] + bias[co];
            }
        }
}

// ---------------------------------------------------------------------------
// Launch
// ---------------------------------------------------------------------------
extern "C" void kernel_launch(void* const* d_in, const int* in_sizes, int n_in,
                              void* d_out, int out_size, void* d_ws, size_t ws_size,
                              hipStream_t stream) {
    const float* xg         = (const float*)d_in[0];  // [16,256,64,64]
    const float* style      = (const float*)d_in[1];  // [16,512]
    const float* mod_weight = (const float*)d_in[2];  // [256,512]
    const float* mod_bias   = (const float*)d_in[3];  // [256]
    const float* weight     = (const float*)d_in[4];  // [1,256,256,3,3]
    const float* bias       = (const float*)d_in[5];  // [1,256,1,1]
    float* out = (float*)d_out;

    // ws: [0,16KB) s ; [16KB, +18.0MB) wmod bf16 ; then x_bf16 (32.0MB)
    float* s_ws = (float*)d_ws;
    unsigned short* wmod = (unsigned short*)((char*)d_ws + 16 * 1024);
    unsigned short* xbf  = (unsigned short*)((char*)d_ws + 16 * 1024
                           + (size_t)BATCH * COUT * NTAP * CIN * 2);

    style_mod_kernel<<<dim3(BATCH), dim3(256), 0, stream>>>(
        style, mod_weight, mod_bias, s_ws);

    demod_wmod_kernel<<<dim3(COUT, BATCH), dim3(256), 0, stream>>>(
        weight, s_ws, wmod);

    x_to_bf16_kernel<<<dim3((BATCH * CIN * NPIX) / (256 * 8)), dim3(256), 0, stream>>>(
        xg, xbf);

    conv_wmma_kernel<<<dim3(NPIX / BT_N, COUT / BT_M, BATCH), dim3(256), 0, stream>>>(
        xbf, wmod, bias, out);
}
